// FineSubMatching_59090160058406
// MI455X (gfx1250) — matrix operand order repaired and checked by reading, not compiled, not run on hardware
//
#include <hip/hip_runtime.h>
#include <hip/hip_bf16.h>

typedef __bf16 bf16;
typedef __attribute__((ext_vector_type(16))) __bf16 v16bf;
typedef __attribute__((ext_vector_type(8)))  __bf16 v8bf;
typedef __attribute__((ext_vector_type(4)))  __bf16 v4bf;
typedef __attribute__((ext_vector_type(8)))  float  v8f;
typedef __attribute__((ext_vector_type(4)))  float  v4f;

#define WW_SZ     64        // fine window elements (8x8)
#define CH        64        // channels
#define LDB       72        // bf16 LDS leading dim (multiple of 8 -> 16B aligned rows)
#define LDS_S     65        // f32 sim LDS leading dim (odd -> bank spread)
#define TEMP_INV  10.0f     // 1 / 0.1
#define INV_SQRT_C 0.125f   // 1 / sqrt(64)
#define THR       0.1f
#define HW_W      80        // coarse grid width (both images)
#define SCALE_F_C 4.0f
#define OUT_SCALE 2.0f
#define HALF_WF   4.0f

// ---- WMMA fragment loaders (bf16, 16x16x32, per CDNA5 ISA layout tables) ----
// A (16x32): lane L holds row m0+L%16; lane-half picks K sub-blocks {0-7,16-23} / {8-15,24-31}.
__device__ __forceinline__ v16bf load_frag_a(const bf16* mat, int m0, int k0, int lane) {
    const int m = lane & 15;
    const int h = (lane >> 4) & 1;
    const bf16* base = mat + (m0 + m) * LDB + k0 + 8 * h;
    v8bf lo = *(const v8bf*)(base);        // K = k0+8h    .. +7
    v8bf hi = *(const v8bf*)(base + 16);   // K = k0+16+8h .. +7
    return __builtin_shufflevector(lo, hi, 0,1,2,3,4,5,6,7,8,9,10,11,12,13,14,15);
}
// B (32x16): lane L holds column n0+L%16; lane-half picks K sub-blocks {0-7,16-23} / {8-15,24-31}.
__device__ __forceinline__ v16bf load_frag_b(const bf16* mat, int k0, int n0, int lane) {
    const int n = lane & 15;
    const int h = (lane >> 4) & 1;
    const bf16* base = mat + (k0 + 8 * h) * LDB + n0 + n;
    v16bf r;
    #pragma unroll
    for (int s = 0; s < 8; ++s) {
        r[s]     = base[s * LDB];
        r[s + 8] = base[(16 + s) * LDB];
    }
    return r;
}

__global__ __launch_bounds__(128)
void xoftr_fine_kernel(const float* __restrict__ f0g, const float* __restrict__ f1g,
                       const float* __restrict__ Wp,  const float* __restrict__ W1,
                       const float* __restrict__ W2,  const int* __restrict__ b_ids,
                       const int* __restrict__ i_ids_c, const int* __restrict__ j_ids_c,
                       float* __restrict__ out, int Mrows)
{
    // 5 bf16 64xLDB tiles = 46080 B; fp32 sim (16640 B) overlays dead sF0/sF1.
    __shared__ char smem_raw[46080] __attribute__((aligned(16)));
    bf16*  sF0  = (bf16*)(smem_raw);
    bf16*  sF1  = (bf16*)(smem_raw + 9216);
    bf16*  sWpT = (bf16*)(smem_raw + 18432);
    bf16*  sP0  = (bf16*)(smem_raw + 27648);   // f0 projected, row-major [l][d]
    bf16*  sP1T = (bf16*)(smem_raw + 36864);   // f1 projected, transposed [d][s]
    float* sSim = (float*)(smem_raw);          // overlays sF0/sF1 after projections

    __shared__ float sRmax[WW_SZ], sRsum[WW_SZ], sCmax[WW_SZ], sCsum[WW_SZ];
    __shared__ float sRedV[128];
    __shared__ int   sRedI[128];
    __shared__ __align__(16) float sX[128];
    __shared__ __align__(16) float sH[128];
    __shared__ float sSub[4];
    __shared__ int   sPeak;
    __shared__ float sPc;

    const int bid  = blockIdx.x;
    const int t    = threadIdx.x;
    const int lane = t & 31;
    const int wave = t >> 5;

    const float* f0w = f0g + (size_t)bid * (WW_SZ * CH);
    const float* f1w = f1g + (size_t)bid * (WW_SZ * CH);

    // ---- Phase 0: stage features + scaled Wp^T into LDS as bf16 (float4 loads) ----
    {
        const v4f* f0v = (const v4f*)f0w;
        const v4f* f1v = (const v4f*)f1w;
        const v4f* wpv = (const v4f*)Wp;
        #pragma unroll
        for (int i = 0; i < 8; ++i) {
            int q  = t + 128 * i;        // float4 index, 0..1023
            int r  = q >> 4;             // row (16 float4 per 64-wide row)
            int c4 = (q & 15) << 2;      // col start
            v4f a = f0v[q];
            v4f b = f1v[q];
            v4f w = wpv[q];
            v4bf ab, bb;
            #pragma unroll
            for (int e = 0; e < 4; ++e) { ab[e] = (bf16)a[e]; bb[e] = (bf16)b[e]; }
            *(v4bf*)&sF0[r * LDB + c4] = ab;     // ds_store_b64
            *(v4bf*)&sF1[r * LDB + c4] = bb;
            // Wp is [d][c] with d=r, c=c4..c4+3; WpT[c][d] = Wp[d][c] * (1/sqrt(C))
            #pragma unroll
            for (int e = 0; e < 4; ++e)
                sWpT[(c4 + e) * LDB + r] = (bf16)(w[e] * INV_SQRT_C);
        }
    }
    __syncthreads();

    // ---- Phase 1: projections f0 = feat0 @ WpT, f1 = feat1 @ WpT ----
    const int m0      = wave * 16;
    const int nn_lane = lane & 15;
    const int hh      = (lane >> 4) & 1;
    #pragma unroll
    for (int which = 0; which < 2; ++which) {
        const bf16* src = which ? sF1 : sF0;
        #pragma unroll
        for (int nt = 0; nt < 4; ++nt) {
            const int n0 = nt * 16;
            v8f acc = {};
            #pragma unroll
            for (int kk = 0; kk < 2; ++kk) {
                v16bf a = load_frag_a(src,  m0, kk * 32, lane);
                v16bf b = load_frag_b(sWpT, kk * 32, n0, lane);
                acc = __builtin_amdgcn_wmma_f32_16x16x32_bf16(
                          false, a, false, b, (short)0, acc, false, false);
            }
            if (which == 0) {
                // row-major: D element (M = m0+8h+r, N = n0+lane%16)
                #pragma unroll
                for (int r = 0; r < 8; ++r)
                    sP0[(m0 + 8 * hh + r) * LDB + n0 + nn_lane] = (bf16)acc[r];
            } else {
                // transposed: sP1T[d][s], d = n0+lane%16, s = m0+8h+r (contig -> ds_store_b128)
                v8bf pk;
                #pragma unroll
                for (int r = 0; r < 8; ++r) pk[r] = (bf16)acc[r];
                *(v8bf*)&sP1T[(n0 + nn_lane) * LDB + m0 + 8 * hh] = pk;
            }
        }
    }
    __syncthreads();

    // ---- Phase 2: sim = (f0 @ f1^T) / TEMP  (fp32 into LDS) ----
    #pragma unroll
    for (int nt = 0; nt < 4; ++nt) {
        const int n0 = nt * 16;
        v8f acc = {};
        #pragma unroll
        for (int kk = 0; kk < 2; ++kk) {
            v16bf a = load_frag_a(sP0,  m0, kk * 32, lane);
            v16bf b = load_frag_b(sP1T, kk * 32, n0, lane);
            acc = __builtin_amdgcn_wmma_f32_16x16x32_bf16(
                      false, a, false, b, (short)0, acc, false, false);
        }
        #pragma unroll
        for (int r = 0; r < 8; ++r)
            sSim[(m0 + 8 * hh + r) * LDS_S + n0 + nn_lane] = acc[r] * TEMP_INV;
    }
    __syncthreads();

    // ---- Phase 3: dual-softmax stats (rows: axis=2, cols: axis=1) ----
    if (t < 64) {
        const float* row = &sSim[t * LDS_S];
        float mx = row[0];
        #pragma unroll 8
        for (int s = 1; s < 64; ++s) mx = fmaxf(mx, row[s]);
        float sm = 0.0f;
        #pragma unroll 8
        for (int s = 0; s < 64; ++s) sm += __expf(row[s] - mx);
        sRmax[t] = mx; sRsum[t] = sm;
    } else {
        const int s = t - 64;
        float mx = sSim[s];
        #pragma unroll 8
        for (int l = 1; l < 64; ++l) mx = fmaxf(mx, sSim[l * LDS_S + s]);
        float sm = 0.0f;
        #pragma unroll 8
        for (int l = 0; l < 64; ++l) sm += __expf(sSim[l * LDS_S + s] - mx);
        sCmax[s] = mx; sCsum[s] = sm;
    }
    __syncthreads();

    // ---- Phase 4: conf = softmax1 * softmax2 (vectorized store) + block argmax ----
    const size_t cbase = (size_t)bid * 4096;
    float bestV = -1.0f; int bestI = 0;
    #pragma unroll
    for (int i = 0; i < 8; ++i) {
        int q     = t + 128 * i;     // float4 chunk, 0..1023
        int flat0 = q << 2;
        int l     = flat0 >> 6;
        int s0    = flat0 & 63;
        float rm = sRmax[l], rsum = sRsum[l];
        v4f cv;
        #pragma unroll
        for (int e = 0; e < 4; ++e) {
            float v = sSim[l * LDS_S + s0 + e];
            float c = __expf(2.0f * v - rm - sCmax[s0 + e]) / (rsum * sCsum[s0 + e]);
            cv[e] = c;
            if (c > bestV) { bestV = c; bestI = flat0 + e; }  // flats ascend -> first-max kept
        }
        *(v4f*)&out[cbase + flat0] = cv;   // global_store_b128, 16B aligned
    }
    sRedV[t] = bestV; sRedI[t] = bestI;
    __syncthreads();
    if (t == 0) {
        float bv = sRedV[0]; int bi = sRedI[0];
        for (int k = 1; k < 128; ++k) {
            float v = sRedV[k]; int ix = sRedI[k];
            if (v > bv || (v == bv && ix < bi)) { bv = v; bi = ix; }
        }
        sPeak = bi; sPc = bv;
    }
    __syncthreads();

    const int peak = sPeak;
    const int i_id = peak >> 6;
    const int j_id = peak & 63;

    // ---- Phase 5: gather peak features + subpixel MLP ----
    sX[t] = (t < 64) ? f0w[i_id * 64 + t] : f1w[j_id * 64 + (t - 64)];
    __syncthreads();
    {
        const v4f* w1row = (const v4f*)(W1 + t * 128);
        const v4f* xv    = (const v4f*)sX;
        float acc = 0.0f;
        #pragma unroll 8
        for (int k = 0; k < 32; ++k) {
            v4f w = w1row[k], x = xv[k];
            acc = fmaf(w[0], x[0], acc);
            acc = fmaf(w[1], x[1], acc);
            acc = fmaf(w[2], x[2], acc);
            acc = fmaf(w[3], x[3], acc);
        }
        sH[t] = fmaxf(acc, 0.0f);
    }
    __syncthreads();
    if (t < 4) {
        const v4f* w2row = (const v4f*)(W2 + t * 128);
        const v4f* hv    = (const v4f*)sH;
        float acc = 0.0f;
        #pragma unroll 8
        for (int k = 0; k < 32; ++k) {
            v4f w = w2row[k], h = hv[k];
            acc = fmaf(w[0], h[0], acc);
            acc = fmaf(w[1], h[1], acc);
            acc = fmaf(w[2], h[2], acc);
            acc = fmaf(w[3], h[3], acc);
        }
        sSub[t] = acc;
    }
    __syncthreads();
    if (t == 0) {
        float s0x = tanhf(sSub[0]) * 0.5f, s0y = tanhf(sSub[1]) * 0.5f;
        float s1x = tanhf(sSub[2]) * 0.5f, s1y = tanhf(sSub[3]) * 0.5f;
        int ic = i_ids_c[bid], jc = j_ids_c[bid];
        float c0x = (float)(ic % HW_W) * SCALE_F_C, c0y = (float)(ic / HW_W) * SCALE_F_C;
        float c1x = (float)(jc % HW_W) * SCALE_F_C, c1y = (float)(jc / HW_W) * SCALE_F_C;
        float w0x = (float)(i_id & 7), w0y = (float)(i_id >> 3);
        float w1x = (float)(j_id & 7), w1y = (float)(j_id >> 3);
        const size_t base0 = (size_t)Mrows * 4096;
        const size_t base1 = base0 + 2 * (size_t)Mrows;
        const size_t base2 = base1 + 2 * (size_t)Mrows;
        const size_t base3 = base2 + (size_t)Mrows;
        out[base0 + 2 * bid + 0] = (w0x + c0x - HALF_WF + s0x) * OUT_SCALE;
        out[base0 + 2 * bid + 1] = (w0y + c0y - HALF_WF + s0y) * OUT_SCALE;
        out[base1 + 2 * bid + 0] = (w1x + c1x - HALF_WF + s1x) * OUT_SCALE;
        out[base1 + 2 * bid + 1] = (w1y + c1y - HALF_WF + s1y) * OUT_SCALE;
        float pc = sPc;
        out[base2 + bid] = (pc > THR) ? pc : 0.0f;
        out[base3 + bid] = (float)b_ids[bid];
    }
}

extern "C" void kernel_launch(void* const* d_in, const int* in_sizes, int n_in,
                              void* d_out, int out_size, void* d_ws, size_t ws_size,
                              hipStream_t stream) {
    const float* f0  = (const float*)d_in[0];
    const float* f1  = (const float*)d_in[1];
    const float* Wp  = (const float*)d_in[2];
    const float* W1  = (const float*)d_in[3];
    const float* W2  = (const float*)d_in[4];
    const int*   bi  = (const int*)d_in[5];
    const int*   ii  = (const int*)d_in[6];
    const int*   ji  = (const int*)d_in[7];
    float* out = (float*)d_out;
    const int Mrows = in_sizes[5];   // one window per coarse match
    xoftr_fine_kernel<<<Mrows, 128, 0, stream>>>(f0, f1, Wp, W1, W2, bi, ii, ji, out, Mrows);
}
